// VGAE_52716428591568
// MI455X (gfx1250) — matrix operand order repaired and checked by reading, not compiled, not run on hardware
//
#include <hip/hip_runtime.h>
#include <hip/hip_bf16.h>

// ---------------------------------------------------------------------------
// VGAE forward for MI455X (gfx1250, wave32, WMMA).
//  - Dense GEMMs: v_wmma_f32_16x16x32_f16, A from b128 loads, B from weights
//    pre-packed once into exact wave32 fragment order (pure b128 loads).
//  - GCN propagation: dinv factored into GEMM stores / epilogues so the
//    dominant scatter kernels are load+atomicAdd only.
//  - Decoder: wave-per-edge dot via cross-lane reduction.
// ---------------------------------------------------------------------------

#define NNODES 100000
#define NEDGES 1600000
#define BAG    10
#define D_EMB  128
#define D_HID  64
#define D_OUT  32

typedef _Float16 v16h __attribute__((ext_vector_type(16)));
typedef _Float16 h2   __attribute__((ext_vector_type(2)));
typedef _Float16 h4   __attribute__((ext_vector_type(4)));
typedef float    v8f  __attribute__((ext_vector_type(8)));

// ---------------- 1) EmbeddingBag(mean): one wave per node ------------------
__global__ void vgae_embed(const int* __restrict__ fidx,
                           const float* __restrict__ table,
                           _Float16* __restrict__ xh) {
  const int wave = (blockIdx.x * blockDim.x + threadIdx.x) >> 5;
  const int lane = threadIdx.x & 31;
  if (wave >= NNODES) return;
  float ax = 0.f, ay = 0.f, az = 0.f, aw = 0.f;
  const int base = wave * BAG;
  #pragma unroll
  for (int j = 0; j < BAG; ++j) {
    const int idx = fidx[base + j];
    const float4 v = ((const float4*)(table + (size_t)idx * D_EMB))[lane];
    ax += v.x; ay += v.y; az += v.z; aw += v.w;
  }
  const float s = 1.0f / (float)BAG;
  h4 o;
  o[0] = (_Float16)(ax * s); o[1] = (_Float16)(ay * s);
  o[2] = (_Float16)(az * s); o[3] = (_Float16)(aw * s);
  *(h4*)(xh + (size_t)wave * D_EMB + lane * 4) = o;
}

// ---------------- 2) degree / norm ------------------------------------------
__global__ void vgae_deg_init(float* __restrict__ deg) {
  const int i = blockIdx.x * blockDim.x + threadIdx.x;
  if (i < NNODES) deg[i] = 1.0f;            // self-loop contributes 1
}
__global__ void vgae_deg_acc(const int* __restrict__ dst, float* __restrict__ deg) {
  const int e = blockIdx.x * blockDim.x + threadIdx.x;
  if (e < NEDGES) atomicAdd(&deg[dst[e]], 1.0f);
}
__global__ void vgae_dinv(float* __restrict__ deg) {
  const int i = blockIdx.x * blockDim.x + threadIdx.x;
  if (i < NNODES) deg[i] = rsqrtf(fmaxf(deg[i], 1.0f));
}

// ---------------- weight pre-pack into WMMA B-fragment order ----------------
// out[((ct*KS + ks)*32 + lane)*16 + t], t -> B[k][col]
// with col = ct*16 + (lane&15), k = ks*32 + (t>>3)*16 + (lane>>4)*8 + (t&7)   (ISA 7.12.2)
__global__ void vgae_packW(const float* __restrict__ W, _Float16* __restrict__ out,
                           int ncols, int ksteps, int total) {
  const int idx = blockIdx.x * blockDim.x + threadIdx.x;
  if (idx >= total) return;
  const int t    = idx & 15;
  const int lane = (idx >> 4) & 31;
  const int fs   = idx >> 9;                  // fragment serial = ct*KS + ks
  const int ks   = fs & (ksteps - 1);
  const int ct   = fs / ksteps;
  const int col  = ct * 16 + (lane & 15);
  const int k    = ks * 32 + ((t & 8) ? 16 : 0) + (lane >> 4) * 8 + (t & 7);   // table-order K
  out[idx] = (_Float16)W[k * ncols + col];
}

// ---------------- 3) GEMM1: (x@W1) * dinv[row] ------------------------------
__global__ void vgae_gemm1(const _Float16* __restrict__ xh,
                           const _Float16* __restrict__ w1p,
                           const float* __restrict__ dinv,
                           float* __restrict__ hlin) {
  const int wave = (blockIdx.x * blockDim.x + threadIdx.x) >> 5;
  const int lane = threadIdx.x & 31;
  if (wave >= NNODES / 16) return;          // wave-uniform: EXEC all-1s in WMMA
  const int m0  = wave * 16;
  const int rlo = lane & 15;
  const int hi  = lane >> 4;

  v16h a[4];
  const _Float16* arow = xh + (size_t)(m0 + rlo) * D_EMB;
  #pragma unroll
  for (int ks = 0; ks < 4; ++ks) {
    #pragma unroll
    for (int v = 0; v < 8; ++v) {
      const int k = ks * 32 + ((v & 4) ? 16 : 0) + hi * 8 + (v & 3) * 2;
      const h2 t = *(const h2*)(arow + k);
      a[ks][2 * v] = t.x; a[ks][2 * v + 1] = t.y;
    }
  }
  float dr[8];
  #pragma unroll
  for (int r = 0; r < 8; ++r) dr[r] = dinv[m0 + hi * 8 + r];

  #pragma unroll
  for (int ct = 0; ct < 4; ++ct) {
    v8f acc = {};
    const int col = ct * 16 + rlo;
    #pragma unroll
    for (int ks = 0; ks < 4; ++ks) {
      const v16h b = *(const v16h*)(w1p + (size_t)((ct * 4 + ks) * 32 + lane) * 16);
      acc = __builtin_amdgcn_wmma_f32_16x16x32_f16(false, a[ks], false, b,
                                                   (short)0, acc, false, false);
    }
    #pragma unroll
    for (int r = 0; r < 8; ++r)
      hlin[(size_t)(m0 + hi * 8 + r) * D_HID + col] = acc[r] * dr[r];
  }
}

// ---------------- 4) propagation layer 1 ------------------------------------
__global__ void vgae_zero(float* __restrict__ p, int n) {
  const int i = blockIdx.x * blockDim.x + threadIdx.x;
  if (i < n) p[i] = 0.0f;
}
__global__ void vgae_prop1(const int* __restrict__ src, const int* __restrict__ dst,
                           const float* __restrict__ hlin,
                           float* __restrict__ hacc) {
  const int gid = blockIdx.x * blockDim.x + threadIdx.x;
  const int e = gid >> 6, c = gid & 63;
  if (e >= NEDGES + NNODES) return;
  int s, d;
  if (e < NEDGES) { s = src[e]; d = dst[e]; } else { s = d = e - NEDGES; }
  atomicAdd(&hacc[(size_t)d * D_HID + c], hlin[(size_t)s * D_HID + c]);
}
__global__ void vgae_post1(const float* __restrict__ dinv, const float* __restrict__ b1,
                           float* __restrict__ hacc) {
  const int i = blockIdx.x * blockDim.x + threadIdx.x;
  if (i < NNODES * D_HID) hacc[i] = dinv[i >> 6] * hacc[i] + b1[i & (D_HID - 1)];
}

// ---------------- 5) GEMM2: (relu(h)@W) * dinv[row], for mu and logstd ------
__global__ void vgae_gemm2(const float* __restrict__ hacc,
                           const _Float16* __restrict__ wmup,
                           const _Float16* __restrict__ wlsp,
                           const float* __restrict__ dinv,
                           float* __restrict__ mulin, float* __restrict__ lslin) {
  const int wave = (blockIdx.x * blockDim.x + threadIdx.x) >> 5;
  const int lane = threadIdx.x & 31;
  if (wave >= NNODES / 16) return;
  const int m0  = wave * 16;
  const int rlo = lane & 15;
  const int hi  = lane >> 4;

  v16h a[2];
  const float* arow = hacc + (size_t)(m0 + rlo) * D_HID;
  #pragma unroll
  for (int ks = 0; ks < 2; ++ks) {
    #pragma unroll
    for (int v = 0; v < 8; ++v) {
      const int k = ks * 32 + ((v & 4) ? 16 : 0) + hi * 8 + (v & 3) * 2;
      const float2 t = *(const float2*)(arow + k);
      a[ks][2 * v]     = (_Float16)fmaxf(t.x, 0.0f);   // fused ReLU
      a[ks][2 * v + 1] = (_Float16)fmaxf(t.y, 0.0f);
    }
  }
  float dr[8];
  #pragma unroll
  for (int r = 0; r < 8; ++r) dr[r] = dinv[m0 + hi * 8 + r];

  #pragma unroll
  for (int mat = 0; mat < 2; ++mat) {
    const _Float16* Wp = mat ? wlsp : wmup;
    float* out         = mat ? lslin : mulin;
    #pragma unroll
    for (int ct = 0; ct < 2; ++ct) {
      v8f acc = {};
      const int col = ct * 16 + rlo;
      #pragma unroll
      for (int ks = 0; ks < 2; ++ks) {
        const v16h b = *(const v16h*)(Wp + (size_t)((ct * 2 + ks) * 32 + lane) * 16);
        acc = __builtin_amdgcn_wmma_f32_16x16x32_f16(false, a[ks], false, b,
                                                     (short)0, acc, false, false);
      }
      #pragma unroll
      for (int r = 0; r < 8; ++r)
        out[(size_t)(m0 + hi * 8 + r) * D_OUT + col] = acc[r] * dr[r];
    }
  }
}

// ---------------- 6) propagation layer 2 (mu & logstd together) -------------
__global__ void vgae_prop2(const int* __restrict__ src, const int* __restrict__ dst,
                           const float* __restrict__ mulin, const float* __restrict__ lslin,
                           float* __restrict__ amu, float* __restrict__ als) {
  const int gid = blockIdx.x * blockDim.x + threadIdx.x;
  const int e = gid >> 5, c = gid & 31;
  if (e >= NEDGES + NNODES) return;
  int s, d;
  if (e < NEDGES) { s = src[e]; d = dst[e]; } else { s = d = e - NEDGES; }
  const size_t so = (size_t)s * D_OUT + c, do_ = (size_t)d * D_OUT + c;
  atomicAdd(&amu[do_], mulin[so]);
  atomicAdd(&als[do_], lslin[so]);
}

// ---------------- 7) epilogue layer 2 + reparameterize ----------------------
__global__ void vgae_reparam(const float* __restrict__ dinv,
                             const float* __restrict__ bmu, const float* __restrict__ bls,
                             const float* __restrict__ amu, const float* __restrict__ als,
                             const float* __restrict__ noise, float* __restrict__ z) {
  const int i = blockIdx.x * blockDim.x + threadIdx.x;
  if (i >= NNODES * D_OUT) return;
  const float dv = dinv[i >> 5];
  const int   c  = i & (D_OUT - 1);
  const float mu = dv * amu[i] + bmu[c];
  const float ls = dv * als[i] + bls[c];
  z[i] = mu + noise[i] * __expf(ls);
}

// ---------------- 8) decoder: wave-per-edge dot + sigmoid -------------------
__global__ void vgae_decode(const int* __restrict__ src, const int* __restrict__ dst,
                            const float* __restrict__ z, float* __restrict__ out) {
  const int wave = (blockIdx.x * blockDim.x + threadIdx.x) >> 5;
  const int lane = threadIdx.x & 31;
  if (wave >= NEDGES) return;
  const int s = src[wave], d = dst[wave];
  float p = z[(size_t)s * D_OUT + lane] * z[(size_t)d * D_OUT + lane];
  #pragma unroll
  for (int off = 16; off; off >>= 1) p += __shfl_xor(p, off, 32);
  if (lane == 0) out[wave] = 1.0f / (1.0f + __expf(-p));
}

// ---------------------------------------------------------------------------
extern "C" void kernel_launch(void* const* d_in, const int* in_sizes, int n_in,
                              void* d_out, int out_size, void* d_ws, size_t ws_size,
                              hipStream_t stream) {
  (void)in_sizes; (void)n_in; (void)out_size; (void)ws_size;
  const int*   fidx  = (const int*)  d_in[0];
  // d_in[1] = feature_offsets: known to be b*BAG, not needed.
  const int*   esrc  = (const int*)  d_in[2];
  const int*   edst  = esrc + NEDGES;
  const float* emb   = (const float*)d_in[3];
  const float* W1    = (const float*)d_in[4];
  const float* b1    = (const float*)d_in[5];
  const float* Wmu   = (const float*)d_in[6];
  const float* bmu   = (const float*)d_in[7];
  const float* Wls   = (const float*)d_in[8];
  const float* bls   = (const float*)d_in[9];
  const float* noise = (const float*)d_in[10];
  float* out = (float*)d_out;

  // Workspace layout (256B-aligned, regions reused across pipeline phases):
  char* ws = (char*)d_ws;
  float*    dinv = (float*)ws;                                   // 400000 B
  _Float16* w1p  = (_Float16*)(ws + 400128);                     // 16384 B
  _Float16* wmup = (_Float16*)(ws + 400128 + 16384);             //  4096 B
  _Float16* wlsp = (_Float16*)(ws + 400128 + 20480);             //  4096 B
  _Float16* xh   = (_Float16*)(ws + 424704);                     // 25.6 MB
  float*    hlin = (float*)(ws + 424704 + 25600000);             // 25.6 MB
  float*    hacc = (float*)(ws + 424704 + 51200000);             // 25.6 MB
  // reuse: hlin region -> mu_lin/ls_lin; xh region -> acc_mu/acc_ls; hacc -> z
  float* mulin = hlin;
  float* lslin = hlin + (size_t)NNODES * D_OUT;
  float* amu   = (float*)xh;
  float* als   = amu + (size_t)NNODES * D_OUT;
  float* z     = hacc;

  const int B = 256;
  // 0) pre-pack weights into WMMA B-fragment order (f16)
  vgae_packW<<<32, B, 0, stream>>>(W1,  w1p,  D_HID, 4, 8192);
  vgae_packW<<< 8, B, 0, stream>>>(Wmu, wmup, D_OUT, 2, 2048);
  vgae_packW<<< 8, B, 0, stream>>>(Wls, wlsp, D_OUT, 2, 2048);
  // 1) embedding bag mean -> xh (f16)
  vgae_embed<<<(NNODES * 32) / B, B, 0, stream>>>(fidx, emb, xh);
  // 2) degrees -> dinv
  vgae_deg_init<<<(NNODES + B - 1) / B, B, 0, stream>>>(dinv);
  vgae_deg_acc<<<NEDGES / B, B, 0, stream>>>(edst, dinv);
  vgae_dinv<<<(NNODES + B - 1) / B, B, 0, stream>>>(dinv);
  // 3) GEMM1 (WMMA), store pre-scaled by dinv[row] -> hlin
  vgae_gemm1<<<(NNODES / 16 * 32 + B - 1) / B, B, 0, stream>>>(xh, w1p, dinv, hlin);
  // 4) propagate layer 1: plain scatter-add, then scale+bias epilogue
  vgae_zero<<<(NNODES * D_HID) / B, B, 0, stream>>>(hacc, NNODES * D_HID);
  vgae_prop1<<<((NEDGES + NNODES) * 64) / B, B, 0, stream>>>(esrc, edst, hlin, hacc);
  vgae_post1<<<(NNODES * D_HID) / B, B, 0, stream>>>(dinv, b1, hacc);
  // 5) GEMM2 (WMMA, fused ReLU), stores pre-scaled by dinv[row]
  vgae_gemm2<<<(NNODES / 16 * 32 + B - 1) / B, B, 0, stream>>>(hacc, wmup, wlsp, dinv,
                                                               mulin, lslin);
  // 6) propagate layer 2: plain scatter-add into amu/als (contiguous region)
  vgae_zero<<<(NNODES * D_HID) / B, B, 0, stream>>>(amu, NNODES * D_HID); // amu+als
  vgae_prop2<<<((NEDGES + NNODES) * 32) / B, B, 0, stream>>>(esrc, edst, mulin, lslin,
                                                             amu, als);
  // 7) epilogue + reparameterize -> z
  vgae_reparam<<<(NNODES * D_OUT) / B, B, 0, stream>>>(dinv, bmu, bls, amu, als, noise, z);
  // 8) decode -> out
  vgae_decode<<<(NEDGES * 32) / B, B, 0, stream>>>(esrc, edst, z, out);
}
